// MSparseSelfAttention_64957085384890
// MI455X (gfx1250) — matrix operand order repaired and checked by reading, not compiled, not run on hardware
//
#include <hip/hip_runtime.h>
#include <hip/hip_bf16.h>

// ---- problem constants ------------------------------------------------------
#define B_      16
#define T_      4096          // power of two: t = m & 4095, b = m >> 12
#define E_      64
#define H_      8
#define KP_     256
#define GADD_   8
#define RADD_   8
#define REGION_ 64
#define SPP_    20            // 4 + GADD + RADD samples per point
#define VS_     5120          // KP_ * SPP_
#define BH_     128           // B_*H_
#define M_      65536         // B_*T_
#define HASHSZ_ 16384         // per-batch dedup hash table (pow2)

typedef __attribute__((ext_vector_type(16))) __bf16 v16bf;
typedef __attribute__((ext_vector_type(8)))  float  v8f;

// ---- float atomic-max via order-preserving int map --------------------------
__device__ __forceinline__ int f2ord(float f) {
  int i = __float_as_int(f);
  return i >= 0 ? i : (i ^ 0x7fffffff);
}
__device__ __forceinline__ float ord2f(int i) {
  return __int_as_float(i >= 0 ? i : (i ^ 0x7fffffff));
}

// ---- init kernels -----------------------------------------------------------
__global__ void fill_i32(int* p, size_t n, int v) {
  for (size_t i = blockIdx.x * (size_t)blockDim.x + threadIdx.x; i < n;
       i += (size_t)gridDim.x * blockDim.x) p[i] = v;
}
__global__ void fill_f32(float* p, size_t n, float v) {
  for (size_t i = blockIdx.x * (size_t)blockDim.x + threadIdx.x; i < n;
       i += (size_t)gridDim.x * blockDim.x) p[i] = v;
}

// ---- 1) per-point params: flipped means -> sigmoid*(t-1), softplus sigma ----
__global__ void point_params(const float* __restrict__ means_p,
                             const float* __restrict__ sigmas_p,
                             float* __restrict__ mx, float* __restrict__ my,
                             float* __restrict__ sg) {
  int k = blockIdx.x * blockDim.x + threadIdx.x;
  if (k >= KP_) return;
  float a = means_p[2 * k + 0], b = means_p[2 * k + 1];
  float hi = fmaxf(a, b), lo = fminf(a, b);
  mx[k] = (1.f / (1.f + __expf(-hi))) * (float)(T_ - 1);
  my[k] = (1.f / (1.f + __expf(-lo))) * (float)(T_ - 1);
  float s = sigmas_p[k] + 2.0f;                              // SIGMA_BOOST
  float sp = (s > 20.f) ? s : __logf(1.f + __expf(s));       // softplus
  sg[k] = (sp + 0.05f) * (float)T_ * 1.0f;                   // MIN_SIGMA, SCALE
}

// ---- 2) index tuple generation (neighbours | global | local), flipped -------
__global__ void build_idx(const float* __restrict__ mx, const float* __restrict__ my,
                          const int* __restrict__ gints, const int* __restrict__ loffs,
                          int* __restrict__ rows, int* __restrict__ cols) {
  int b = blockIdx.y;
  int j = blockIdx.x * blockDim.x + threadIdx.x;
  if (j >= VS_) return;
  int k = j / SPP_, slot = j % SPP_;
  float m0 = mx[k], m1 = my[k];
  int r, c;
  if (slot < 4) {
    float p0 = (float)((slot >> 1) & 1), p1 = (float)(slot & 1);
    float i0 = floorf(m0) * (1.f - p0) + ceilf(m0) * p0;
    float i1 = floorf(m1) * (1.f - p1) + ceilf(m1) * p1;
    r = (int)fminf(fmaxf(i0, 0.f), (float)(T_ - 1));
    c = (int)fminf(fmaxf(i1, 0.f), (float)(T_ - 1));
  } else if (slot < 4 + GADD_) {
    const int* p = gints + (((size_t)(b * KP_ + k) * GADD_ + (slot - 4)) * 2);
    r = p[0]; c = p[1];
  } else {
    int l0 = (int)fminf(fmaxf(floorf(m0) - REGION_ / 2, 0.f), (float)(T_ - REGION_));
    int l1 = (int)fminf(fmaxf(floorf(m1) - REGION_ / 2, 0.f), (float)(T_ - REGION_));
    const int* p = loffs + (((size_t)(b * KP_ + k) * RADD_ + (slot - 4 - GADD_)) * 2);
    r = l0 + p[0]; c = l1 + p[1];
  }
  int hi = max(r, c), lo = min(r, c);                         // flip: row >= col
  rows[b * VS_ + j] = hi;
  cols[b * VS_ + j] = lo;
}

// ---- 3) duplicate marking via per-batch hash table --------------------------
__global__ void dedup_k(const int* __restrict__ rows, const int* __restrict__ cols,
                        int* __restrict__ hash, int* __restrict__ dup) {
  int b = blockIdx.y;
  int j = blockIdx.x * blockDim.x + threadIdx.x;
  if (j >= VS_) return;
  int id = rows[b * VS_ + j] * T_ + cols[b * VS_ + j];
  unsigned h = ((unsigned)id * 2654435761u) & (HASHSZ_ - 1);
  int* tab = hash + (size_t)b * HASHSZ_;
  int d = 0;
  for (int probe = 0; probe < HASHSZ_; ++probe) {
    int prev = atomicCAS(&tab[h], -1, id);
    if (prev == -1) { d = 0; break; }           // first claimer keeps weight
    if (prev == id) { d = 1; break; }           // duplicate -> zeroed
    h = (h + 1) & (HASHSZ_ - 1);
  }
  dup[b * VS_ + j] = d;
}

// ---- 4) per-(b,k) column sums of unnormalized Gaussian densities ------------
__global__ void colsum_k(const int* __restrict__ rows, const int* __restrict__ cols,
                         const int* __restrict__ dup,
                         const float* __restrict__ mx, const float* __restrict__ my,
                         const float* __restrict__ sg, float* __restrict__ colsum) {
  int k = blockIdx.x, b = blockIdx.y;
  float m0 = mx[k], m1 = my[k], inv = 1.f / sg[k];
  float acc = 0.f;
  for (int j = threadIdx.x; j < VS_; j += blockDim.x) {
    if (dup[b * VS_ + j]) continue;
    float d0 = ((float)rows[b * VS_ + j] - m0) * inv;
    float d1 = ((float)cols[b * VS_ + j] - m1) * inv;
    acc += __expf(-0.5f * (d0 * d0 + d1 * d1));
  }
  __shared__ float red[256];
  red[threadIdx.x] = acc;
  __syncthreads();
  for (int s = 128; s > 0; s >>= 1) {
    if (threadIdx.x < s) red[threadIdx.x] += red[threadIdx.x + s];
    __syncthreads();
  }
  if (threadIdx.x == 0) colsum[b * KP_ + k] = red[0];
}

// ---- 5) per-edge weight = sum_k props/colsum --------------------------------
__global__ void weights_k(const int* __restrict__ rows, const int* __restrict__ cols,
                          const int* __restrict__ dup,
                          const float* __restrict__ mx, const float* __restrict__ my,
                          const float* __restrict__ sg, const float* __restrict__ colsum,
                          float* __restrict__ wts) {
  int b = blockIdx.y;
  int j = blockIdx.x * blockDim.x + threadIdx.x;
  if (j >= VS_) return;
  if (dup[b * VS_ + j]) { wts[b * VS_ + j] = 0.f; return; }
  float r = (float)rows[b * VS_ + j], c = (float)cols[b * VS_ + j];
  float acc = 0.f;
  for (int k = 0; k < KP_; ++k) {
    float inv = 1.f / sg[k];
    float d0 = (r - mx[k]) * inv, d1 = (c - my[k]) * inv;
    float cs = colsum[b * KP_ + k];
    acc += (cs > 0.f) ? (__expf(-0.5f * (d0 * d0 + d1 * d1)) / cs) : 0.f;
  }
  wts[b * VS_ + j] = acc;
}

// ---- 6) QKV projection GEMM: (65536x64)*(64x512), WMMA bf16 -----------------
// Output written directly in (b*H+h, t, e) head-major layout, pre-scaled.
__global__ __launch_bounds__(256) void gemm_qkv(const float* __restrict__ x,
                                                const float* __restrict__ w,
                                                float* __restrict__ out, float scale) {
  int wave = threadIdx.x >> 5, lane = threadIdx.x & 31;
  int half = lane >> 4, ml = lane & 15;
  int mtile = blockIdx.x * 16;
  int ntile = blockIdx.y * 128 + wave * 16;

  __shared__ __bf16 xa[16][64];                 // A tile staged via LDS as bf16
  for (int i = threadIdx.x; i < 16 * 64; i += 256) {
    int r = i >> 6, cc = i & 63;
    xa[r][cc] = (__bf16)x[(size_t)(mtile + r) * 64 + cc];
  }
  __syncthreads();

  v8f c = {};
#pragma unroll
  for (int ks = 0; ks < 2; ++ks) {
    int kb = ks * 32;
    __builtin_prefetch(&w[(size_t)(kb + half * 16) * 512 + ntile + ml], 0, 0);
    v16bf a, bm;
#pragma unroll
    for (int i = 0; i < 8; ++i) {               // ISA 16-bit A 16x32 layout
      a[i]     = xa[ml][kb + half * 8 + i];
      a[8 + i] = xa[ml][kb + 16 + half * 8 + i];
    }
#pragma unroll
    for (int i = 0; i < 8; ++i) {               // B: K striped across halves
      bm[i]     = (__bf16)w[(size_t)(kb + half * 8 + i) * 512 + ntile + ml];
      bm[8 + i] = (__bf16)w[(size_t)(kb + 16 + half * 8 + i) * 512 + ntile + ml];
    }
    c = __builtin_amdgcn_wmma_f32_16x16x32_bf16(false, a, false, bm,
                                                (short)0, c, false, false);
  }
#pragma unroll
  for (int r = 0; r < 8; ++r) {                 // C/D layout: m = half*8 + r
    int m = mtile + half * 8 + r;
    int n = ntile + ml;
    int b = m >> 12, t = m & (T_ - 1);
    int hh = n >> 6, e = n & 63;
    out[((size_t)(b * H_ + hh) * T_ + t) * E_ + e] = c[r] * scale;
  }
}

// ---- 7) per-edge q.k dot (weighted) + running row max -----------------------
__global__ void edge_dot(const float* __restrict__ q, const float* __restrict__ kk,
                         const int* __restrict__ rows, const int* __restrict__ cols,
                         const float* __restrict__ wts,
                         float* __restrict__ dotv, int* __restrict__ rowmaxi) {
  size_t idx = blockIdx.x * (size_t)blockDim.x + threadIdx.x;
  if (idx >= (size_t)BH_ * VS_) return;
  int bh = (int)(idx / VS_), j = (int)(idx % VS_);
  int b = bh >> 3;
  int r = rows[b * VS_ + j], c = cols[b * VS_ + j];
  float w = wts[b * VS_ + j];
  const float4* qp = (const float4*)(q + ((size_t)bh * T_ + r) * E_);
  const float4* kp = (const float4*)(kk + ((size_t)bh * T_ + c) * E_);
  float acc = 0.f;
#pragma unroll
  for (int i = 0; i < 16; ++i) {
    float4 a = qp[i], bb = kp[i];
    acc += a.x * bb.x + a.y * bb.y + a.z * bb.z + a.w * bb.w;
  }
  float d = w * acc;
  dotv[idx] = d;
  atomicMax(&rowmaxi[(size_t)bh * T_ + r], f2ord(d));
}

// ---- 8) exp(dot - rowmax), accumulate row sums ------------------------------
__global__ void edge_exp(const float* __restrict__ dotv, const int* __restrict__ rows,
                         const int* __restrict__ rowmaxi,
                         float* __restrict__ exv, float* __restrict__ rowsum) {
  size_t idx = blockIdx.x * (size_t)blockDim.x + threadIdx.x;
  if (idx >= (size_t)BH_ * VS_) return;
  int bh = (int)(idx / VS_), j = (int)(idx % VS_);
  int b = bh >> 3;
  int r = rows[b * VS_ + j];
  float m = ord2f(rowmaxi[(size_t)bh * T_ + r]);
  float e = __expf(dotv[idx] - m);
  exv[idx] = e;
  atomicAdd(&rowsum[(size_t)bh * T_ + r], e);
}

// ---- 9) scatter-add  ex * V[col] -> acc[row]  (acc is L2-resident, 134MB) ---
__global__ void edge_scatter(const float* __restrict__ exv, const float* __restrict__ v,
                             const int* __restrict__ rows, const int* __restrict__ cols,
                             float* __restrict__ acc) {
  size_t gid = blockIdx.x * (size_t)blockDim.x + threadIdx.x;
  if (gid >= (size_t)BH_ * VS_ * 16) return;
  int chunk = (int)(gid & 15);
  size_t eidx = gid >> 4;
  int bh = (int)(eidx / VS_), j = (int)(eidx % VS_);
  int b = bh >> 3;
  float e = exv[eidx];
  int r = rows[b * VS_ + j], c = cols[b * VS_ + j];
  float4 vv = *(const float4*)(v + ((size_t)bh * T_ + c) * E_ + chunk * 4);
  float* ap = acc + ((size_t)bh * T_ + r) * E_ + chunk * 4;
  atomicAdd(ap + 0, e * vv.x);
  atomicAdd(ap + 1, e * vv.y);
  atomicAdd(ap + 2, e * vv.z);
  atomicAdd(ap + 3, e * vv.w);
}

// ---- 10) output GEMM: (65536x512)*(512x64)+bu, WMMA bf16 --------------------
// Softmax normalization (acc/rowsum, 0 if empty row) folded into A-operand load.
__device__ __forceinline__ float ctx_elem(const float* acc, const float* rowsum,
                                          int b, int t, int n) {
  int hh = n >> 6, ee = n & 63;
  size_t row = (size_t)(b * H_ + hh) * T_ + t;
  float rs = rowsum[row];
  return rs > 0.f ? acc[row * E_ + ee] / rs : 0.f;
}

__global__ __launch_bounds__(256) void gemm_out(const float* __restrict__ acc,
                                                const float* __restrict__ rowsum,
                                                const float* __restrict__ Wu,
                                                const float* __restrict__ bu,
                                                float* __restrict__ y) {
  int wave = threadIdx.x >> 5, lane = threadIdx.x & 31;
  int half = lane >> 4, ml = lane & 15;
  int mtile = blockIdx.x * 32 + (wave >> 2) * 16;
  int ntile = (wave & 3) * 16;

  v8f c = {};
  int m = mtile + ml;
  int b = m >> 12, t = m & (T_ - 1);
#pragma unroll 4
  for (int ks = 0; ks < 16; ++ks) {
    int kb = ks * 32;
    __builtin_prefetch(&Wu[(size_t)(kb + 32 + half * 16) * 64 + ntile + ml], 0, 0);
    v16bf a, bm;
#pragma unroll
    for (int i = 0; i < 8; ++i) {
      a[i]     = (__bf16)ctx_elem(acc, rowsum, b, t, kb + half * 8 + i);
      a[8 + i] = (__bf16)ctx_elem(acc, rowsum, b, t, kb + 16 + half * 8 + i);
    }
#pragma unroll
    for (int i = 0; i < 8; ++i) {
      bm[i]     = (__bf16)Wu[(size_t)(kb + half * 8 + i) * 64 + ntile + ml];
      bm[8 + i] = (__bf16)Wu[(size_t)(kb + 16 + half * 8 + i) * 64 + ntile + ml];
    }
    c = __builtin_amdgcn_wmma_f32_16x16x32_bf16(false, a, false, bm,
                                                (short)0, c, false, false);
  }
#pragma unroll
  for (int r = 0; r < 8; ++r) {
    int mm = mtile + half * 8 + r;
    int n = ntile + ml;
    y[(size_t)mm * E_ + n] = c[r] + bu[n];
  }
}

// ---- host orchestration -----------------------------------------------------
extern "C" void kernel_launch(void* const* d_in, const int* in_sizes, int n_in,
                              void* d_out, int out_size, void* d_ws, size_t ws_size,
                              hipStream_t stream) {
  (void)in_sizes; (void)n_in; (void)out_size; (void)ws_size;
  const float* x        = (const float*)d_in[0];
  const float* Wq       = (const float*)d_in[1];
  const float* Wk       = (const float*)d_in[2];
  const float* Wv       = (const float*)d_in[3];
  const float* Wu       = (const float*)d_in[4];
  const float* bu       = (const float*)d_in[5];
  const float* means_p  = (const float*)d_in[6];
  const float* sigmas_p = (const float*)d_in[7];
  const int*   gints    = (const int*)d_in[8];
  const int*   loffs    = (const int*)d_in[9];
  float* y = (float*)d_out;

  char* wsp = (char*)d_ws;
  auto alloc = [&](size_t bytes) -> char* {
    char* p = wsp;
    wsp += (bytes + 255) & ~(size_t)255;
    return p;
  };

  int*   rows   = (int*)  alloc((size_t)B_ * VS_ * 4);
  int*   cols   = (int*)  alloc((size_t)B_ * VS_ * 4);
  int*   dup    = (int*)  alloc((size_t)B_ * VS_ * 4);
  int*   hash   = (int*)  alloc((size_t)B_ * HASHSZ_ * 4);
  float* wts    = (float*)alloc((size_t)B_ * VS_ * 4);
  float* mx     = (float*)alloc((size_t)KP_ * 4);
  float* my     = (float*)alloc((size_t)KP_ * 4);
  float* sg     = (float*)alloc((size_t)KP_ * 4);
  float* colsum = (float*)alloc((size_t)B_ * KP_ * 4);
  float* qb     = (float*)alloc((size_t)BH_ * T_ * E_ * 4);
  float* kb     = (float*)alloc((size_t)BH_ * T_ * E_ * 4);
  float* vb     = (float*)alloc((size_t)BH_ * T_ * E_ * 4);
  float* accb   = (float*)alloc((size_t)BH_ * T_ * E_ * 4);
  int*   rmaxi  = (int*)  alloc((size_t)BH_ * T_ * 4);
  float* rsum   = (float*)alloc((size_t)BH_ * T_ * 4);
  float* dotv   = (float*)alloc((size_t)BH_ * VS_ * 4);
  float* exv    = (float*)alloc((size_t)BH_ * VS_ * 4);

  const size_t nAcc  = (size_t)BH_ * T_ * E_;
  const size_t nRow  = (size_t)BH_ * T_;
  const size_t nEdge = (size_t)BH_ * VS_;
  const float qscale = 0.35355339059327373f;   // E^-0.25, E=64

  // init
  fill_i32<<<2048, 256, 0, stream>>>(hash,  (size_t)B_ * HASHSZ_, -1);
  fill_i32<<<2048, 256, 0, stream>>>(rmaxi, nRow, INT_MIN);
  fill_f32<<<2048, 256, 0, stream>>>(rsum,  nRow, 0.f);
  fill_f32<<<8192, 256, 0, stream>>>(accb,  nAcc, 0.f);

  // sparse pattern + weights
  point_params<<<dim3((KP_ + 255) / 256), 256, 0, stream>>>(means_p, sigmas_p, mx, my, sg);
  build_idx<<<dim3((VS_ + 255) / 256, B_), 256, 0, stream>>>(mx, my, gints, loffs, rows, cols);
  dedup_k<<<dim3((VS_ + 255) / 256, B_), 256, 0, stream>>>(rows, cols, hash, dup);
  colsum_k<<<dim3(KP_, B_), 256, 0, stream>>>(rows, cols, dup, mx, my, sg, colsum);
  weights_k<<<dim3((VS_ + 255) / 256, B_), 256, 0, stream>>>(rows, cols, dup, mx, my, sg,
                                                             colsum, wts);

  // Q/K/V projections (WMMA bf16), written head-major, pre-scaled
  dim3 gq(M_ / 16, 512 / 128);
  gemm_qkv<<<gq, 256, 0, stream>>>(x, Wq, qb, qscale);
  gemm_qkv<<<gq, 256, 0, stream>>>(x, Wk, kb, qscale);
  gemm_qkv<<<gq, 256, 0, stream>>>(x, Wv, vb, 1.0f);

  // sparse attention
  edge_dot<<<(unsigned)((nEdge + 255) / 256), 256, 0, stream>>>(qb, kb, rows, cols, wts,
                                                                dotv, rmaxi);
  edge_exp<<<(unsigned)((nEdge + 255) / 256), 256, 0, stream>>>(dotv, rows, rmaxi, exv, rsum);
  edge_scatter<<<(unsigned)((nEdge * 16 + 255) / 256), 256, 0, stream>>>(exv, vb, rows, cols,
                                                                         accb);

  // output projection (normalization folded in) — WMMA bf16
  gemm_out<<<dim3(M_ / 32), 256, 0, stream>>>(accb, rsum, Wu, bu, y);
}
